// Attention_45526653337996
// MI455X (gfx1250) — compile-verified, matrix operand order checked by reading, not run on hardware
//
#include <hip/hip_runtime.h>
#include <math.h>

#define B_     8
#define HH     56
#define WW_    56
#define N_     3136
#define DIM_   512
#define SD0    64
#define CL     448
#define HEADS_ 8
#define DH     28
#define WIN_   7
#define W2     49
#define LK     784
#define NTOP   40
#define NS     25
#define SCALE0 0.125f
#define SCALE_L 0.1889822365046136f  /* 28^-0.5 */

#define CDIV(a,b) (((a)+(b)-1)/(b))

typedef __attribute__((ext_vector_type(16))) _Float16 v16h;
typedef __attribute__((ext_vector_type(8)))  float    v8f;

// ---------------------------------------------------------------------------
// WMMA GEMM: C[M,N] = A[M,K] @ B[K,N] (+bias), fp32 in/out, f16 matrix MACs.
// One wave -> TM stacked 16x16 tiles: each strided B tile is reused by TM
// WMMAs; A loads are contiguous b128. Grid tiles M/N exactly (no guards, so
// EXEC stays all-ones and accumulators never move). NN (compile-time) folds
// all strided B/C addressing into immediate offsets.
// BM: 0=no bias, 1=per-column, 2=per-row. Batch via blockIdx.z strides.
// ---------------------------------------------------------------------------
template <int BM, int TM, int NN>
__global__ void wmma_gemm_f16(const float* __restrict__ A, const float* __restrict__ Bmat,
                              const float* __restrict__ bias, float* __restrict__ C,
                              int K, long sA, long sB, long sC)
{
    A    += (long)blockIdx.z * sA;
    Bmat += (long)blockIdx.z * sB;
    C    += (long)blockIdx.z * sC;
    const int lane   = threadIdx.x & 31;
    const int wave   = threadIdx.x >> 5;
    const int tmBase = (blockIdx.y * 4 + wave) * TM;   // wave-uniform, exact tiling
    const int col    = blockIdx.x * 16 + (lane & 15);
    const int kh     = (lane >> 4) * 8;                // K-half select (16-bit layout)
    const int rowi   = lane & 15;

    v8f acc[TM];
#pragma unroll
    for (int t = 0; t < TM; ++t)
#pragma unroll
        for (int v = 0; v < 8; ++v) acc[t][v] = 0.f;

    for (int k0 = 0; k0 < K; k0 += 32) {
        v16h b;
        const float* bp = Bmat + (long)(k0 + kh) * NN + col;
#pragma unroll
        for (int j = 0; j < 8; ++j) {
            b[j]     = (_Float16)bp[j * NN];            // immediate offsets
            b[j + 8] = (_Float16)bp[(j + 16) * NN];
        }
#pragma unroll
        for (int t = 0; t < TM; ++t) {
            const float* ap = A + (long)((tmBase + t) * 16 + rowi) * K + k0 + kh;
            v16h a;
#pragma unroll
            for (int j = 0; j < 8; ++j) {
                a[j]     = (_Float16)ap[j];             // K = k0+kh+j
                a[j + 8] = (_Float16)ap[j + 16];        // K = k0+16+kh+j
            }
            acc[t] = __builtin_amdgcn_wmma_f32_16x16x32_f16(false, a, false, b,
                                                            (short)0, acc[t], false, false);
        }
    }

    const float bcol = (BM == 1) ? bias[col] : 0.f;
#pragma unroll
    for (int t = 0; t < TM; ++t) {
        const int m0 = (tmBase + t) * 16 + 8 * (lane >> 4);
        float* cp = C + (long)m0 * NN + col;
#pragma unroll
        for (int v = 0; v < 8; ++v) {
            float o = acc[t][v];
            if (BM == 1) o += bcol;
            if (BM == 2) o += bias[m0 + v];
            cp[v * NN] = o;                             // immediate offsets
        }
    }
}

// ---------------------------------------------------------------------------
// sr conv: 2x2 stride-2, 64->64ch. in[b,c,h,w] = x[b, h*56+w, c]. out xg[b,p,o].
// ---------------------------------------------------------------------------
__global__ void sr_conv(const float* __restrict__ x, const float* __restrict__ wgt,
                        const float* __restrict__ bias, float* __restrict__ xg)
{
    int idx = blockIdx.x * blockDim.x + threadIdx.x;
    if (idx >= B_ * LK * SD0) return;
    int o = idx % SD0;
    int p = (idx / SD0) % LK;
    int b = idx / (SD0 * LK);
    int i = p / 28, j = p % 28;
    float acc = bias[o];
    for (int c = 0; c < SD0; ++c) {
#pragma unroll
        for (int ky = 0; ky < 2; ++ky)
#pragma unroll
            for (int kx = 0; kx < 2; ++kx) {
                int h = 2 * i + ky, ww = 2 * j + kx;
                acc += wgt[(o * SD0 + c) * 4 + ky * 2 + kx] *
                       x[((long)b * N_ + h * WW_ + ww) * DIM_ + c];
            }
    }
    xg[((long)b * LK + p) * SD0 + o] = acc;
}

__global__ void ln64(const float* __restrict__ in, const float* __restrict__ g,
                     const float* __restrict__ bb, float* __restrict__ outp)
{
    int idx = blockIdx.x * blockDim.x + threadIdx.x;
    if (idx >= B_ * LK) return;
    const float* r = in + (long)idx * SD0;
    float m = 0.f;
    for (int e = 0; e < SD0; ++e) m += r[e];
    m *= (1.f / SD0);
    float v = 0.f;
    for (int e = 0; e < SD0; ++e) { float d = r[e] - m; v += d * d; }
    v *= (1.f / SD0);
    float rs = rsqrtf(v + 1e-5f);
    float* o = outp + (long)idx * SD0;
    for (int e = 0; e < SD0; ++e) o[e] = (r[e] - m) * rs * g[e] + bb[e];
}

// ---------------------------------------------------------------------------
// Global-branch depthwise conv with flat-reinterpret layout:
// vr[b,c,p] = v_g[b, (c*784+p)//64, (c*784+p)%64];  v2[b,l,e] = v_g + conv[e, l]
// ---------------------------------------------------------------------------
__global__ void lcg_conv(const float* __restrict__ kv, const float* __restrict__ lw,
                         const float* __restrict__ lb, float* __restrict__ v2)
{
    int idx = blockIdx.x * blockDim.x + threadIdx.x;
    if (idx >= B_ * LK * SD0) return;
    int e = idx % SD0;
    int l = (idx / SD0) % LK;
    int b = idx / (SD0 * LK);
    int i = l / 28, j = l % 28;
    float acc = lb[e];
#pragma unroll
    for (int ky = 0; ky < 3; ++ky) {
        int ii = i + ky - 1;
        if (ii < 0 || ii >= 28) continue;
#pragma unroll
        for (int kx = 0; kx < 3; ++kx) {
            int jj = j + kx - 1;
            if (jj < 0 || jj >= 28) continue;
            int F = e * LK + ii * 28 + jj;
            acc += lw[e * 9 + ky * 3 + kx] *
                   kv[((long)b * LK + (F >> 6)) * 128 + 64 + (F & 63)];
        }
    }
    v2[idx] = kv[((long)b * LK + l) * 128 + 64 + e] + acc;
}

// M[b,n] = max_s(QK) - mean_s(QK), 25 sampled keys, unscaled.
__global__ void mscore(const float* __restrict__ q, const float* __restrict__ kv,
                       const int* __restrict__ ridx, float* __restrict__ Mout)
{
    int idx = blockIdx.x * blockDim.x + threadIdx.x;
    if (idx >= B_ * N_) return;
    int b = idx / N_, n = idx % N_;
    const float* qp = q + ((long)b * N_ + n) * DIM_;
    float mx = -INFINITY, sum = 0.f;
    for (int s = 0; s < NS; ++s) {
        int l = ridx[n * NS + s];
        const float* kp = kv + ((long)b * LK + l) * 128;
        float d = 0.f;
        for (int e = 0; e < SD0; ++e) d += qp[e] * kp[e];
        mx = fmaxf(mx, d); sum += d;
    }
    Mout[idx] = mx - sum * (1.f / NS);
}

// top-40 per batch, value desc / index asc tie-break (matches lax.top_k).
__global__ void topk40(const float* __restrict__ M, int* __restrict__ Mtop)
{
    int b = blockIdx.x;
    __shared__ float sm[N_];
    __shared__ float rv[256];
    __shared__ int   ri[256];
    for (int i = threadIdx.x; i < N_; i += 256) sm[i] = M[(long)b * N_ + i];
    __syncthreads();
    for (int t = 0; t < NTOP; ++t) {
        float bv = -INFINITY; int bi = 0x7fffffff;
        for (int i = threadIdx.x; i < N_; i += 256) {
            float v = sm[i];
            if (v > bv || (v == bv && i < bi)) { bv = v; bi = i; }
        }
        rv[threadIdx.x] = bv; ri[threadIdx.x] = bi;
        __syncthreads();
        for (int s = 128; s > 0; s >>= 1) {
            if (threadIdx.x < s) {
                float ov = rv[threadIdx.x + s]; int oi = ri[threadIdx.x + s];
                if (ov > rv[threadIdx.x] ||
                    (ov == rv[threadIdx.x] && oi < ri[threadIdx.x])) {
                    rv[threadIdx.x] = ov; ri[threadIdx.x] = oi;
                }
            }
            __syncthreads();
        }
        if (threadIdx.x == 0) { Mtop[b * NTOP + t] = ri[0]; sm[ri[0]] = -INFINITY; }
        __syncthreads();
    }
}

__global__ void glogits(const float* __restrict__ q, const float* __restrict__ kv,
                        const int* __restrict__ Mtop, float* __restrict__ lg)
{
    int idx = blockIdx.x * blockDim.x + threadIdx.x;
    if (idx >= B_ * NTOP * LK) return;
    int w = idx % LK;
    int t = (idx / LK) % NTOP;
    int b = idx / (LK * NTOP);
    int n = Mtop[b * NTOP + t];
    const float* qp = q + ((long)b * N_ + n) * DIM_;
    const float* kp = kv + ((long)b * LK + w) * 128;
    float d = 0.f;
    for (int e = 0; e < SD0; ++e) d += qp[e] * kp[e];
    lg[idx] = d * SCALE0;
}

__global__ void gsoftmax(float* __restrict__ lg)
{
    float* row = lg + (long)blockIdx.x * LK;
    __shared__ float red[256];
    float mx = -INFINITY;
    for (int i = threadIdx.x; i < LK; i += 256) mx = fmaxf(mx, row[i]);
    red[threadIdx.x] = mx; __syncthreads();
    for (int s = 128; s > 0; s >>= 1) {
        if (threadIdx.x < s) red[threadIdx.x] = fmaxf(red[threadIdx.x], red[threadIdx.x + s]);
        __syncthreads();
    }
    mx = red[0]; __syncthreads();
    float sum = 0.f;
    for (int i = threadIdx.x; i < LK; i += 256) { float e = __expf(row[i] - mx); row[i] = e; sum += e; }
    red[threadIdx.x] = sum; __syncthreads();
    for (int s = 128; s > 0; s >>= 1) {
        if (threadIdx.x < s) red[threadIdx.x] += red[threadIdx.x + s];
        __syncthreads();
    }
    float inv = 1.f / red[0];
    for (int i = threadIdx.x; i < LK; i += 256) row[i] *= inv;
}

__global__ void gvsel(const float* __restrict__ lg, const float* __restrict__ v2,
                      float* __restrict__ vsel)
{
    int idx = blockIdx.x * blockDim.x + threadIdx.x;
    if (idx >= B_ * NTOP * SD0) return;
    int e = idx % SD0;
    int t = (idx / SD0) % NTOP;
    int b = idx / (SD0 * NTOP);
    const float* lr = lg + ((long)b * NTOP + t) * LK;
    float acc = 0.f;
    for (int w = 0; w < LK; ++w) acc += lr[w] * v2[((long)b * LK + w) * SD0 + e];
    vsel[idx] = acc;
}

__global__ void zerocat64(float* __restrict__ cat)
{
    int idx = blockIdx.x * blockDim.x + threadIdx.x;
    if (idx >= B_ * N_ * SD0) return;
    int e = idx % SD0;
    long bn = idx / SD0;
    cat[bn * DIM_ + e] = 0.f;
}

__global__ void gscatter(const float* __restrict__ vsel, const int* __restrict__ Mtop,
                         float* __restrict__ cat)
{
    int idx = blockIdx.x * blockDim.x + threadIdx.x;
    if (idx >= B_ * NTOP * SD0) return;
    int e = idx % SD0;
    int t = (idx / SD0) % NTOP;
    int b = idx / (SD0 * NTOP);
    int n = Mtop[b * NTOP + t];
    cat[((long)b * N_ + n) * DIM_ + e] = vsel[idx];
}

// depthwise 3x3 pad1, 448ch, in from x[:, :, 64:], out kv_int[b,c,p] (NCHW-flat)
__global__ void dw_conv(const float* __restrict__ x, const float* __restrict__ wgt,
                        const float* __restrict__ bias, float* __restrict__ outp)
{
    long idx = (long)blockIdx.x * blockDim.x + threadIdx.x;
    if (idx >= (long)B_ * CL * N_) return;
    int p = (int)(idx % N_);
    int c = (int)((idx / N_) % CL);
    int b = (int)(idx / ((long)N_ * CL));
    int i = p / WW_, j = p % WW_;
    float acc = bias[c];
#pragma unroll
    for (int ky = 0; ky < 3; ++ky) {
        int ii = i + ky - 1;
        if (ii < 0 || ii >= HH) continue;
#pragma unroll
        for (int kx = 0; kx < 3; ++kx) {
            int jj = j + kx - 1;
            if (jj < 0 || jj >= WW_) continue;
            acc += wgt[c * 9 + ky * 3 + kx] *
                   x[((long)b * N_ + ii * WW_ + jj) * DIM_ + SD0 + c];
        }
    }
    outp[idx] = acc;
}

// kvp[b,h,wi,j] = mean_{l'} kv_l[b, ch0+h*28+(F%28), pix(F//28, wi)], F=j*L+l'
__global__ void kvp_kernel(const float* __restrict__ kvl, float* __restrict__ kvp,
                           int br, int Lc, int gw)
{
    int idx = blockIdx.x * blockDim.x + threadIdx.x;
    if (idx >= B_ * HEADS_ * W2 * DH) return;
    int j  = idx % DH;
    int wi = (idx / DH) % W2;
    int h  = (idx / (DH * W2)) % HEADS_;
    int b  = idx / (DH * W2 * HEADS_);
    int ky = wi / WIN_, kx = wi % WIN_;
    float acc = 0.f;
    for (int lp = 0; lp < Lc; ++lp) {
        int F = j * Lc + lp;
        int l = F / DH, dd = F % DH;
        int gi = l / gw, gj = l % gw;
        int hp, wp;
        if (br == 0) { hp = gi * 7 + ky; wp = gj * 7 + kx; }
        else {
            hp = gi * 13 + ky * 2; wp = gj * 13 + kx * 2;  // padded 65x65
            if (hp > 55) hp = 110 - hp;                    // reflect
            if (wp > 55) wp = 110 - wp;
        }
        acc += kvl[((long)b * CL + br * 224 + h * DH + dd) * N_ + hp * WW_ + wp];
    }
    kvp[idx] = acc / (float)Lc;
}

__global__ void mm_fc(const float* __restrict__ in, const float* __restrict__ fc,
                      float* __restrict__ outp)
{
    int idx = blockIdx.x * blockDim.x + threadIdx.x;
    if (idx >= B_ * HEADS_ * W2 * DH) return;
    int j = idx % DH;
    const float* r = in + (idx - j);
    float acc = 0.f;
    for (int i = 0; i < DH; ++i) acc += r[i] * fc[i * DH + j];
    outp[idx] = acc;
}

__global__ void mm_sh(const float* __restrict__ in, const float* __restrict__ sh,
                      float* __restrict__ outp)
{
    int idx = blockIdx.x * blockDim.x + threadIdx.x;
    if (idx >= B_ * HEADS_ * W2 * 56) return;
    int j = idx % 56;
    const float* r = in + (long)(idx / 56) * DH;
    float acc = 0.f;
    for (int i = 0; i < DH; ++i) acc += r[i] * sh[i * 56 + j];
    outp[idx] = acc;
}

// branch depthwise 3x3 pad1 over the 7x7 window grid; v0f = v + conv(v)+bias
__global__ void lc_branch(const float* __restrict__ t2, const float* __restrict__ lw,
                          const float* __restrict__ lb, float* __restrict__ v0f)
{
    int idx = blockIdx.x * blockDim.x + threadIdx.x;
    if (idx >= B_ * HEADS_ * W2 * DH) return;
    int dd = idx % DH;
    int wi = (idx / DH) % W2;
    int h  = (idx / (DH * W2)) % HEADS_;
    int b  = idx / (DH * W2 * HEADS_);
    int ch = h * DH + dd;
    int i = wi / WIN_, j = wi % WIN_;
    long base = ((long)(b * HEADS_ + h) * W2) * 56;
    float acc = lb[ch];
#pragma unroll
    for (int ky = 0; ky < 3; ++ky) {
        int ii = i + ky - 1;
        if (ii < 0 || ii >= WIN_) continue;
#pragma unroll
        for (int kx = 0; kx < 3; ++kx) {
            int jj = j + kx - 1;
            if (jj < 0 || jj >= WIN_) continue;
            acc += lw[ch * 9 + ky * 3 + kx] * t2[base + (ii * WIN_ + jj) * 56 + DH + dd];
        }
    }
    v0f[idx] = t2[base + wi * 56 + DH + dd] + acc;
}

// fused local attention: per (b,h) stage K(49x28) & V in LDS, each thread = one n.
__global__ void branch_attn(const float* __restrict__ q, const float* __restrict__ t2,
                            const float* __restrict__ vf, float* __restrict__ cat,
                            int qoff, int choff)
{
    int bh = blockIdx.y;
    int b = bh >> 3, h = bh & 7;
    int n = blockIdx.x * 256 + threadIdx.x;
    __shared__ float sk[W2 * DH];
    __shared__ float sv[W2 * DH];
    const float* t2b = t2 + (long)bh * W2 * 56;
    const float* vfb = vf + (long)bh * W2 * DH;
    for (int i = threadIdx.x; i < W2 * DH; i += 256) {
        int w = i / DH, dd = i % DH;
        sk[i] = t2b[w * 56 + dd];
        sv[i] = vfb[i];
    }
    __syncthreads();
    if (n >= N_) return;
    const float* qp = q + ((long)b * N_ + n) * DIM_ + SD0 + h * 56 + qoff;
    float qv[DH];
#pragma unroll
    for (int dd = 0; dd < DH; ++dd) qv[dd] = qp[dd];
    float mxd = -INFINITY;
    for (int w = 0; w < W2; ++w) {
        float s = 0.f;
#pragma unroll
        for (int dd = 0; dd < DH; ++dd) s += qv[dd] * sk[w * DH + dd];
        mxd = fmaxf(mxd, s);
    }
    float sum = 0.f;
    float acc[DH];
#pragma unroll
    for (int dd = 0; dd < DH; ++dd) acc[dd] = 0.f;
    for (int w = 0; w < W2; ++w) {
        float s = 0.f;
#pragma unroll
        for (int dd = 0; dd < DH; ++dd) s += qv[dd] * sk[w * DH + dd];
        float p = __expf((s - mxd) * SCALE_L);
        sum += p;
#pragma unroll
        for (int dd = 0; dd < DH; ++dd) acc[dd] += p * sv[w * DH + dd];
    }
    float inv = 1.f / sum;
    float* op = cat + ((long)b * N_ + n) * DIM_ + SD0 + choff + h * DH;
#pragma unroll
    for (int dd = 0; dd < DH; ++dd) op[dd] = acc[dd] * inv;
}

// ---------------------------------------------------------------------------
extern "C" void kernel_launch(void* const* d_in, const int* in_sizes, int n_in,
                              void* d_out, int out_size, void* d_ws, size_t ws_size,
                              hipStream_t stream)
{
    const float* x      = (const float*)d_in[0];
    const float* q_w    = (const float*)d_in[1];
    const float* kv_g_w = (const float*)d_in[2];
    const float* sr_w   = (const float*)d_in[3];
    const float* sr_b   = (const float*)d_in[4];
    const float* norm_g = (const float*)d_in[5];
    const float* norm_b = (const float*)d_in[6];
    const float* lcg_w  = (const float*)d_in[7];
    const float* lcg_b  = (const float*)d_in[8];
    const float* dw_w   = (const float*)d_in[9];
    const float* dw_b   = (const float*)d_in[10];
    const float* pw_w   = (const float*)d_in[11];
    const float* pw_b   = (const float*)d_in[12];
    const float* fcw[2] = {(const float*)d_in[13], (const float*)d_in[17]};
    const float* shw[2] = {(const float*)d_in[14], (const float*)d_in[18]};
    const float* lcw[2] = {(const float*)d_in[15], (const float*)d_in[19]};
    const float* lcb[2] = {(const float*)d_in[16], (const float*)d_in[20]};
    const float* proj_w = (const float*)d_in[21];
    const float* proj_b = (const float*)d_in[22];
    const int*   ridx   = (const int*)d_in[23];
    float* out = (float*)d_out;
    (void)in_sizes; (void)n_in; (void)out_size; (void)ws_size;

    float* w = (float*)d_ws;
    size_t off = 0;
    auto alloc = [&](size_t n) { float* p = w + off; off += n; return p; };
    float* q     = alloc((size_t)B_ * N_ * DIM_);
    float* cat   = alloc((size_t)B_ * N_ * DIM_);
    float* kvint = alloc((size_t)B_ * CL * N_);
    float* kvl   = alloc((size_t)B_ * CL * N_);
    float* xg    = alloc((size_t)B_ * LK * SD0);
    float* xgln  = alloc((size_t)B_ * LK * SD0);
    float* kvg   = alloc((size_t)B_ * LK * 2 * SD0);
    float* vg2   = alloc((size_t)B_ * LK * SD0);
    float* Mbuf  = alloc((size_t)B_ * N_);
    int*   Mtop  = (int*)alloc(512);
    float* glog  = alloc((size_t)B_ * NTOP * LK);
    float* vsel  = alloc((size_t)B_ * NTOP * SD0);
    float* kvp   = alloc((size_t)B_ * HEADS_ * W2 * DH);
    float* t1    = alloc((size_t)B_ * HEADS_ * W2 * DH);
    float* t2    = alloc((size_t)B_ * HEADS_ * W2 * 56);
    float* v0f   = alloc((size_t)B_ * HEADS_ * W2 * DH);

    // 1) q = x @ q_w   (WMMA: 25088 x 512 x 512; 1568 M-tiles = 98 blocks x 4w x TM4)
    wmma_gemm_f16<0, 4, DIM_><<<dim3(DIM_ / 16, 98, 1), 128, 0, stream>>>(
        x, q_w, nullptr, q, DIM_, 0, 0, 0);

    // 2) global branch prep
    sr_conv<<<CDIV(B_ * LK * SD0, 256), 256, 0, stream>>>(x, sr_w, sr_b, xg);
    ln64<<<CDIV(B_ * LK, 256), 256, 0, stream>>>(xg, norm_g, norm_b, xgln);
    // kv = xgln @ kv_g_w   (WMMA: 6272 x 128 x 64; 392 M-tiles = 49 x 4w x TM2)
    wmma_gemm_f16<0, 2, 128><<<dim3(128 / 16, 49, 1), 128, 0, stream>>>(
        xgln, kv_g_w, nullptr, kvg, SD0, 0, 0, 0);
    lcg_conv<<<CDIV(B_ * LK * SD0, 256), 256, 0, stream>>>(kvg, lcg_w, lcg_b, vg2);

    // 3) sampled scores + top-40 + sparse attention + scatter
    mscore<<<CDIV(B_ * N_, 256), 256, 0, stream>>>(q, kvg, ridx, Mbuf);
    topk40<<<B_, 256, 0, stream>>>(Mbuf, Mtop);
    glogits<<<CDIV(B_ * NTOP * LK, 256), 256, 0, stream>>>(q, kvg, Mtop, glog);
    gsoftmax<<<B_ * NTOP, 256, 0, stream>>>(glog);
    gvsel<<<CDIV(B_ * NTOP * SD0, 256), 256, 0, stream>>>(glog, vg2, vsel);
    zerocat64<<<CDIV(B_ * N_ * SD0, 256), 256, 0, stream>>>(cat);
    gscatter<<<CDIV(B_ * NTOP * SD0, 256), 256, 0, stream>>>(vsel, Mtop, cat);

    // 4) local kv: depthwise 3x3 then pointwise
    //    (WMMA GEMM 448 x 3136 x 448, x8 batch; 28 M-tiles = 7 blocks x 4w x TM1)
    dw_conv<<<(int)CDIV((long)B_ * CL * N_, 256), 256, 0, stream>>>(x, dw_w, dw_b, kvint);
    wmma_gemm_f16<2, 1, N_><<<dim3(N_ / 16, 7, B_), 128, 0, stream>>>(
        pw_w, kvint, pw_b, kvl, CL, 0, (long)CL * N_, (long)CL * N_);

    // 5) two window-attention branches (buffers reused sequentially)
    const int Lc[2] = {64, 25};
    const int gw[2] = {8, 5};
    for (int br = 0; br < 2; ++br) {
        kvp_kernel<<<CDIV(B_ * HEADS_ * W2 * DH, 256), 256, 0, stream>>>(
            kvl, kvp, br, Lc[br], gw[br]);
        mm_fc<<<CDIV(B_ * HEADS_ * W2 * DH, 256), 256, 0, stream>>>(kvp, fcw[br], t1);
        mm_sh<<<CDIV(B_ * HEADS_ * W2 * 56, 256), 256, 0, stream>>>(t1, shw[br], t2);
        lc_branch<<<CDIV(B_ * HEADS_ * W2 * DH, 256), 256, 0, stream>>>(
            t2, lcw[br], lcb[br], v0f);
        branch_attn<<<dim3(CDIV(N_, 256), B_ * HEADS_), 256, 0, stream>>>(
            q, t2, v0f, cat, br * DH, br * 224);
    }

    // 6) out = cat @ proj_w + proj_b   (WMMA: 25088 x 512 x 512)
    wmma_gemm_f16<1, 4, DIM_><<<dim3(DIM_ / 16, 98, 1), 128, 0, stream>>>(
        cat, proj_w, proj_b, out, DIM_, 0, 0, 0);
}
